// MSAI_807453851824
// MI455X (gfx1250) — compile-verified, matrix operand order checked by reading, not run on hardware
//
#include <hip/hip_runtime.h>

typedef __attribute__((ext_vector_type(16))) __bf16 v16bf;
typedef __attribute__((ext_vector_type(8)))  float  v8f;
typedef __attribute__((ext_vector_type(4)))  unsigned int v4u;
typedef __attribute__((ext_vector_type(8)))  int v8i;
typedef __attribute__((ext_vector_type(4)))  int v4i;

#define BB 8
#define CC 64
#define HH 128
#define WW 128
#define HWSZ (HH * WW)
#define NE (BB * CC * HWSZ)   // 8,388,608 elements per tensor
#define BN 128                // N-tile = one image row

// ---- CDNA5 data-mover feature gates (never break the build) ----------------
#if defined(__AMDGCN__) && __has_builtin(__builtin_amdgcn_tensor_load_to_lds) && \
    __has_builtin(__builtin_amdgcn_s_wait_tensorcnt)
#define HAVE_TDM 1
#else
#define HAVE_TDM 0
#endif

#if defined(__AMDGCN__) && __has_builtin(__builtin_amdgcn_global_load_async_to_lds_b128) && \
    __has_builtin(__builtin_amdgcn_s_wait_asynccnt)
#define HAVE_ASYNC 1
#else
#define HAVE_ASYNC 0
#endif

__device__ __forceinline__ v8f wmma_bf16(v16bf a, v16bf b, v8f c) {
    // D = A(16x32 bf16) x B(32x16 bf16) + C(16x16 f32)
    return __builtin_amdgcn_wmma_f32_16x16x32_bf16(
        /*neg_a=*/false, a, /*neg_b=*/false, b,
        /*c_mod=*/(short)0, c, /*reuse_a=*/false, /*reuse_b=*/false);
}

// ---------------------------------------------------------------------------
// Stage an 8 KB contiguous bf16 weight tile (64x64) global -> LDS.
// Tier 1: Tensor Data Mover (one tensor_load_to_lds from wave 0, TENSORcnt).
// Tier 2: async load-to-LDS (ASYNCcnt). Tier 3: plain VALU copy.
// Caller must __syncthreads() before consuming.
// ---------------------------------------------------------------------------
__device__ __forceinline__ void stage_A(const __bf16* __restrict__ src,
                                        __bf16* dst_lds, int tid, int wid)
{
#if HAVE_TDM
    if (wid == 0) {
        const unsigned long long ga = (unsigned long long)(const void*)src;
        v4u g0;
        g0.x = 1u;                                    // count=1, user, no gather
        g0.y = (unsigned)(unsigned long long)dst_lds; // LDS byte offset (flat lo32)
        g0.z = (unsigned)(ga & 0xffffffffu);          // global_addr[31:0]
        g0.w = (unsigned)((ga >> 32) & 0x1ffffffu)    // global_addr[56:32]
             | (2u << 30);                            // type=2 ("image")
        // group1: data_size=2B, 1-D tensor: dim0=4096, tile_dim0=4096, stride0=4096
        v8i g1 = { 0x00010000,            // workgroup_mask=0, data_size=1(2B)
                   (int)(4096u << 16),    // tensor_dim0 low16 at bits 63:48
                   0,                     // tensor_dim0 hi16 / tensor_dim1 lo16
                   (int)(4096u << 16),    // tile_dim0 at bits 127:112
                   0,                     // tile_dim1=0 (unused), tile_dim2=0
                   4096,                  // tensor_dim0_stride low32
                   0, 0 };
#if __has_include(<hip/amd_detail/amd_gfx1250_TDM.h>)
        __builtin_amdgcn_tensor_load_to_lds(g0, g1, (v4i)0, (v4i)0, (v8i)0, 0);
#else
        __builtin_amdgcn_tensor_load_to_lds(g0, g1, (v4i)0, (v4i)0, 0);
#endif
        __builtin_amdgcn_s_wait_tensorcnt(0);
    }
#elif HAVE_ASYNC
    typedef __attribute__((address_space(1))) void gvoid;
    typedef __attribute__((address_space(3))) void lvoid;
    for (int i = tid; i < 512; i += 256)
        __builtin_amdgcn_global_load_async_to_lds_b128(
            (gvoid*)((const int4*)src + i), (lvoid*)((int4*)dst_lds + i), 0, 0);
    __builtin_amdgcn_s_wait_asynccnt(0);
#else
    {
        const int4* s4 = (const int4*)src;
        int4* d4 = (int4*)dst_lds;
        for (int i = tid; i < 512; i += 256) d4[i] = s4[i];
    }
#endif
}

// ---------------------------------------------------------------------------
// Generic implicit-GEMM conv kernel.
//   out[b, m, h, n] (+)= sum_{tap, c} W[tap][m][c] * in[b, c, h+dy*dil, n+dx*dil]
// M = 64 (all output channels), N = 128 (one image row), K = 64 per tap.
// Block: 256 threads = 8 waves; wave grid 4(M) x 2(N); each wave: 16x64 tile
// as four 16x16 f32 WMMA accumulators.
// ---------------------------------------------------------------------------
template<bool IN_F32, bool RELU_IN, int NTAPS, bool ACC_OUT, bool OUT_BF16, bool RELU_OUT>
__global__ __launch_bounds__(256) void conv_gemm(
    const void* __restrict__ inv, const __bf16* __restrict__ Wq,
    const float* __restrict__ bias, void* __restrict__ outv, int dil)
{
    __shared__ alignas(32) __bf16 As[CC * 64];    // [m][k]  64x64  (8 KB)
    __shared__ alignas(32) __bf16 Bst[BN * 64];   // [n][k] 128x64 (16 KB), transposed

    const int blk  = blockIdx.x;          // B * (HWSZ/BN) = 1024 blocks
    const int b    = blk >> 7;
    const int h    = blk & 127;
    const int tid  = threadIdx.x;
    const int lane = tid & 31;
    const int wid  = tid >> 5;
    const int m0   = (wid >> 1) << 4;     // 0,16,32,48
    const int nb   = (wid & 1) << 6;      // 0,64

    const float*  inF = (const float*)inv;
    const __bf16* inH = (const __bf16*)inv;

    v8f acc[4] = {};

    for (int tap = 0; tap < NTAPS; ++tap) {
        const int dy = (NTAPS == 9) ? ((tap / 3) - 1) * dil : 0;
        const int dx = (NTAPS == 9) ? ((tap % 3) - 1) * dil : 0;

        // ---- stage A tile (TDM / async / copy) ----
        stage_A(Wq + (size_t)tap * 4096, As, tid, wid);

        // ---- stage B tile transposed: Bst[px][c] with tap shift + pad ----
        const int  hs  = h + dy;
        const bool hok = (unsigned)hs < (unsigned)HH;
        for (int i = tid; i < CC * BN; i += 256) {
            const int c   = i >> 7;
            const int px  = i & 127;
            const int wsx = px + dx;
            __bf16 hv = (__bf16)0.f;
            if (hok && (unsigned)wsx < (unsigned)WW) {
                const size_t idx = ((size_t)(b * CC + c)) * HWSZ + (size_t)hs * WW + wsx;
                if (IN_F32) {
                    float v = inF[idx];
                    if (RELU_IN) v = fmaxf(v, 0.f);
                    hv = (__bf16)v;
                } else {
                    hv = inH[idx];
                }
            }
            Bst[px * 64 + c] = hv;
        }

        // ---- prefetch next tap's row into L2 (global_prefetch_b8) ----
        if (NTAPS == 9 && tap + 1 < NTAPS) {
            const int ndy = ((tap + 1) / 3 - 1) * dil;
            const int nhs = h + ndy;
            if ((unsigned)nhs < (unsigned)HH) {
                const int c = tid >> 2;
                const size_t idx = ((size_t)(b * CC + c)) * HWSZ + (size_t)nhs * WW + ((tid & 3) << 5);
                __builtin_prefetch(IN_F32 ? (const void*)(inF + idx)
                                          : (const void*)(inH + idx), 0, 1);
            }
        }
        __syncthreads();

        // ---- batched fragment loads, then 8 back-to-back v_wmma ----
        {
            const int arow = (m0 + (lane & 15)) * 64 + ((lane >> 4) << 4);
            const v16bf a0 = *(const v16bf*)&As[arow];
            const v16bf a1 = *(const v16bf*)&As[arow + 32];
            v16bf b0[4], b1[4];
#pragma unroll
            for (int s = 0; s < 4; ++s) {
                const int brow = (nb + s * 16 + (lane & 15)) * 64 + ((lane >> 4) << 4);
                b0[s] = *(const v16bf*)&Bst[brow];
                b1[s] = *(const v16bf*)&Bst[brow + 32];
            }
#pragma unroll
            for (int s = 0; s < 4; ++s) acc[s] = wmma_bf16(a0, b0[s], acc[s]);
#pragma unroll
            for (int s = 0; s < 4; ++s) acc[s] = wmma_bf16(a1, b1[s], acc[s]);
        }
        __syncthreads();
    }

    // ---- epilogue: bias, optional accumulate / relu / bf16 store ----
    float*  outF = (float*)outv;
    __bf16* outH = (__bf16*)outv;
    const int mlo = (lane >> 4) << 3;
    const int nlo = lane & 15;
#pragma unroll
    for (int s = 0; s < 4; ++s) {
        const int col = nb + s * 16 + nlo;
#pragma unroll
        for (int r = 0; r < 8; ++r) {
            const int m = m0 + mlo + r;
            float v = acc[s][r];
            if (bias) v += bias[m];
            const size_t o = ((size_t)(b * CC + m)) * HWSZ + (size_t)h * WW + col;
            if (ACC_OUT)            outF[o] += v;
            else if (OUT_BF16)      outH[o]  = (__bf16)(RELU_OUT ? fmaxf(v, 0.f) : v);
            else                    outF[o]  = RELU_OUT ? fmaxf(v, 0.f) : v;
        }
    }
}

// ---------------------------------------------------------------------------
// Depthwise 3x3 conv (groups = C), pad 1: tiny FLOP count -> VALU.
// ---------------------------------------------------------------------------
__global__ __launch_bounds__(256) void depthwise3x3(
    const __bf16* __restrict__ in, const float* __restrict__ w,
    const float* __restrict__ bias, __bf16* __restrict__ out)
{
    const size_t i  = (size_t)blockIdx.x * 256 + threadIdx.x;
    const int px = (int)(i & (HWSZ - 1));
    const int c  = (int)((i >> 14) & 63);
    const int h  = px >> 7, x = px & 127;
    const size_t base = i - px;           // start of this (b,c) plane
    float s = bias[c];
#pragma unroll
    for (int t = 0; t < 9; ++t) {
        const int hy = h + t / 3 - 1;
        const int wx = x + t % 3 - 1;
        if ((unsigned)hy < (unsigned)HH && (unsigned)wx < (unsigned)WW)
            s += w[c * 9 + t] * (float)in[base + hy * WW + wx];
    }
    out[i] = (__bf16)s;
}

// ---------------------------------------------------------------------------
// Involution core: patches -> per-channel 9x9 matvec -> BN -> softmax ->
// product tile (K padded 9->16 per channel) in LDS -> K=1024 WMMA contraction
// against zero-padded conv_w. Block: 128-pixel row, all 64 outputs.
// ---------------------------------------------------------------------------
__global__ __launch_bounds__(256) void involution_kernel(
    const float* __restrict__ y2, const float* __restrict__ gw_w,
    const float* __restrict__ gscale, const float* __restrict__ gshift,
    const __bf16* __restrict__ Wp, const float* __restrict__ conv_b,
    __bf16* __restrict__ out3)
{
    __shared__ alignas(32) __bf16 Pst[BN * 128];  // [px][k-chunk], 32 KB

    const int blk  = blockIdx.x;
    const int b    = blk >> 7;
    const int h    = blk & 127;
    const int tid  = threadIdx.x;
    const int lane = tid & 31;
    const int wid  = tid >> 5;
    const int m0   = (wid >> 1) << 4;
    const int nb   = (wid & 1) << 6;

    v8f acc[4] = {};

    for (int cc8 = 0; cc8 < 8; ++cc8) {           // 8 chunks of 8 channels
        __syncthreads();
        for (int it = 0; it < 4; ++it) {          // 1024 (c,px) pairs / 256 thr
            const int idx = tid + (it << 8);
            const int cl  = idx >> 7;
            const int px  = idx & 127;
            const int c   = (cc8 << 3) + cl;
            const float* base = y2 + ((size_t)(b * CC + c)) * HWSZ;

            float patch[9];
#pragma unroll
            for (int t = 0; t < 9; ++t) {
                const int hy = h + t / 3 - 1, wx = px + t % 3 - 1;
                patch[t] = ((unsigned)hy < (unsigned)HH && (unsigned)wx < (unsigned)WW)
                               ? base[hy * WW + wx] : 0.f;
            }
            const float* gw = gw_w + c * 81;
            float g[9], mx = -1e30f;
#pragma unroll
            for (int o = 0; o < 9; ++o) {
                float a = 0.f;
#pragma unroll
                for (int q = 0; q < 9; ++q) a += gw[o * 9 + q] * patch[q];
                a = a * gscale[c * 9 + o] + gshift[c * 9 + o];
                g[o] = a;
                mx = fmaxf(mx, a);
            }
            float se = 0.f;
#pragma unroll
            for (int o = 0; o < 9; ++o) { g[o] = __expf(g[o] - mx); se += g[o]; }
            const float inv = 1.f / se;

            __bf16* dst = &Pst[px * 128 + (cl << 4)];
#pragma unroll
            for (int k = 0; k < 9; ++k)  dst[k] = (__bf16)(patch[k] * g[k] * inv);
#pragma unroll
            for (int k = 9; k < 16; ++k) dst[k] = (__bf16)0.f;
        }
        __syncthreads();

#pragma unroll
        for (int kc = 0; kc < 4; ++kc) {          // 4 K-steps of 32 per chunk
            const v16bf afrag = *(const v16bf*)
                (Wp + (size_t)(m0 + (lane & 15)) * 1024 + (cc8 << 7)
                    + kc * 32 + ((lane >> 4) << 4));
            v16bf bf[4];
#pragma unroll
            for (int s = 0; s < 4; ++s)
                bf[s] = *(const v16bf*)
                    &Pst[(nb + s * 16 + (lane & 15)) * 128 + kc * 32 + ((lane >> 4) << 4)];
#pragma unroll
            for (int s = 0; s < 4; ++s) acc[s] = wmma_bf16(afrag, bf[s], acc[s]);
        }
    }

    const int mlo = (lane >> 4) << 3;
    const int nlo = lane & 15;
#pragma unroll
    for (int s = 0; s < 4; ++s) {
        const int col = nb + s * 16 + nlo;
#pragma unroll
        for (int r = 0; r < 8; ++r) {
            const int m = m0 + mlo + r;
            const float v = acc[s][r] + conv_b[m];
            out3[((size_t)(b * CC + m)) * HWSZ + (size_t)h * WW + col] = (__bf16)v;
        }
    }
}

// ---------------------------------------------------------------------------
// Weight prep: bf16 conversion, tap-major re-layout, BN/ensemble/identity folds.
// ---------------------------------------------------------------------------
__global__ __launch_bounds__(256) void prep_weights(
    const float* conv1_w, const float* w5, const float* w7, const float* w9,
    const float* w11, const float* pw_w,
    const float* gw_b, const float* gbn_g, const float* gbn_b,
    const float* gbn_m, const float* gbn_v,
    const float* conv_w, const float* fr_w,
    const float* ens_w, const float* ens_b,
    const float* bn_g, const float* bn_b, const float* bn_m, const float* bn_v,
    __bf16* W_conv1, __bf16* W_dil, __bf16* W_pw,
    float* gscale, float* gshift, __bf16* Wp,
    __bf16* W_fr, __bf16* W_ens, float* bias_f)
{
    const int tid = threadIdx.x;
    for (int i = tid; i < 4096; i += 256) {
        W_conv1[i] = (__bf16)conv1_w[i];
        W_pw[i]    = (__bf16)pw_w[i];
        // feature reuse: conv1x1(x) + x  ==  (fr_w + I) x
        W_fr[i]    = (__bf16)(fr_w[i] + (((i >> 6) == (i & 63)) ? 1.f : 0.f));
    }
    const float* wd[4] = {w5, w7, w9, w11};
    for (int br = 0; br < 4; ++br)
        for (int i = tid; i < 9 * 4096; i += 256) {
            const int tap = i >> 12, co = (i >> 6) & 63, ci = i & 63;
            W_dil[(size_t)br * 9 * 4096 + i] =
                (__bf16)wd[br][((co << 6) + ci) * 9 + tap];
        }
    for (int i = tid; i < 576; i += 256) {
        const float s = gbn_g[i] * rsqrtf(gbn_v[i] + 1e-5f);
        gscale[i] = s;
        gshift[i] = (gw_b[i] - gbn_m[i]) * s + gbn_b[i];
    }
    for (int i = tid; i < 64 * 1024; i += 256) {   // conv_w padded 9 -> 16 per c
        const int o = i >> 10, c = (i >> 4) & 63, k = i & 15;
        Wp[i] = (k < 9) ? (__bf16)conv_w[((o << 6) + c) * 9 + k] : (__bf16)0.f;
    }
    for (int i = tid; i < 9 * 4096; i += 256) {    // ensemble mean + BN fold
        const int tap = i >> 12, o = (i >> 6) & 63, m = i & 63;
        const float em = (ens_w[(size_t)(((0   + o) << 6) + m) * 9 + tap]
                        + ens_w[(size_t)(((64  + o) << 6) + m) * 9 + tap]
                        + ens_w[(size_t)(((128 + o) << 6) + m) * 9 + tap]) * (1.f / 3.f);
        const float s = bn_g[o] * rsqrtf(bn_v[o] + 1e-5f);
        W_ens[i] = (__bf16)(s * em);
    }
    for (int i = tid; i < 64; i += 256) {
        const float s = bn_g[i] * rsqrtf(bn_v[i] + 1e-5f);
        bias_f[i] = s * ((ens_b[i] + ens_b[64 + i] + ens_b[128 + i]) * (1.f / 3.f)
                         - bn_m[i]) + bn_b[i];
    }
}

__global__ __launch_bounds__(256) void zero_f32(float4* __restrict__ p, int n4) {
    const int i = blockIdx.x * 256 + threadIdx.x;
    if (i < n4) p[i] = make_float4(0.f, 0.f, 0.f, 0.f);
}

// ---------------------------------------------------------------------------
extern "C" void kernel_launch(void* const* d_in, const int* in_sizes, int n_in,
                              void* d_out, int out_size, void* d_ws, size_t ws_size,
                              hipStream_t stream)
{
    const float* x       = (const float*)d_in[0];
    const float* conv1_w = (const float*)d_in[1];
    const float* conv1_b = (const float*)d_in[2];
    const float* w5      = (const float*)d_in[3];
    const float* w7      = (const float*)d_in[4];
    const float* w9      = (const float*)d_in[5];
    const float* w11     = (const float*)d_in[6];
    const float* dw_w    = (const float*)d_in[7];
    const float* dw_b    = (const float*)d_in[8];
    const float* pw_w    = (const float*)d_in[9];
    const float* pw_b    = (const float*)d_in[10];
    const float* gw_w    = (const float*)d_in[11];
    const float* gw_b    = (const float*)d_in[12];
    const float* gbn_g   = (const float*)d_in[13];
    const float* gbn_b   = (const float*)d_in[14];
    const float* gbn_m   = (const float*)d_in[15];
    const float* gbn_v   = (const float*)d_in[16];
    const float* conv_w  = (const float*)d_in[17];
    const float* conv_b  = (const float*)d_in[18];
    const float* fr_w    = (const float*)d_in[19];
    const float* fr_b    = (const float*)d_in[20];
    const float* ens_w   = (const float*)d_in[21];
    const float* ens_b   = (const float*)d_in[22];
    const float* bn_g    = (const float*)d_in[23];
    const float* bn_b    = (const float*)d_in[24];
    const float* bn_m    = (const float*)d_in[25];
    const float* bn_v    = (const float*)d_in[26];

    char* base = (char*)d_ws;
    size_t off = 0;
    auto alloc = [&](size_t bytes) -> void* {
        void* p = base + off;
        off = (off + bytes + 255) & ~(size_t)255;
        return p;
    };
    __bf16* W_conv1 = (__bf16*)alloc(4096 * 2);
    __bf16* W_dil   = (__bf16*)alloc((size_t)4 * 9 * 4096 * 2);
    __bf16* W_pw    = (__bf16*)alloc(4096 * 2);
    float*  gscale  = (float*) alloc(576 * 4);
    float*  gshift  = (float*) alloc(576 * 4);
    __bf16* Wp      = (__bf16*)alloc((size_t)64 * 1024 * 2);
    __bf16* W_fr    = (__bf16*)alloc(4096 * 2);
    __bf16* W_ens   = (__bf16*)alloc((size_t)9 * 4096 * 2);
    float*  bias_f  = (float*) alloc(64 * 4);
    __bf16* y1      = (__bf16*)alloc((size_t)NE * 2);  // conv1 output
    __bf16* t1      = (__bf16*)alloc((size_t)NE * 2);  // dilated conv output
    __bf16* t2      = (__bf16*)alloc((size_t)NE * 2);  // depthwise output
    float*  y2      = (float*) alloc((size_t)NE * 4);  // CDCM branch accumulator
    __bf16* out3    = (__bf16*)alloc((size_t)NE * 2);  // involution output
    __bf16* hbuf    = (__bf16*)alloc((size_t)NE * 2);  // feature-reuse output

    const dim3 blk(256);
    const dim3 ggemm(BB * (HWSZ / BN));   // 1024

    prep_weights<<<1, blk, 0, stream>>>(conv1_w, w5, w7, w9, w11, pw_w,
        gw_b, gbn_g, gbn_b, gbn_m, gbn_v, conv_w, fr_w, ens_w, ens_b,
        bn_g, bn_b, bn_m, bn_v,
        W_conv1, W_dil, W_pw, gscale, gshift, Wp, W_fr, W_ens, bias_f);

    zero_f32<<<NE / 4 / 256, blk, 0, stream>>>((float4*)y2, NE / 4);

    // y1 = conv1x1(relu(x)) + b
    conv_gemm<true, true, 1, false, true, false>
        <<<ggemm, blk, 0, stream>>>(x, W_conv1, conv1_b, y1, 1);

    // CDCM: 4 x [dilated 3x3 -> depthwise 3x3 -> pointwise 1x1], summed in y2
    const int dils[4] = {5, 7, 9, 11};
    for (int br = 0; br < 4; ++br) {
        conv_gemm<false, false, 9, false, true, false>
            <<<ggemm, blk, 0, stream>>>(y1, W_dil + (size_t)br * 9 * 4096,
                                        nullptr, t1, dils[br]);
        depthwise3x3<<<NE / 256, blk, 0, stream>>>(t1, dw_w, dw_b, t2);
        conv_gemm<false, false, 1, true, false, false>
            <<<ggemm, blk, 0, stream>>>(t2, W_pw, pw_b, y2, 1);
    }

    // involution core (patches, gw, softmax, K=576 contraction)
    involution_kernel<<<ggemm, blk, 0, stream>>>(y2, gw_w, gscale, gshift,
                                                 Wp, conv_b, out3);

    // feature reuse: (fr_w + I) * out3 + fr_b
    conv_gemm<false, false, 1, false, true, false>
        <<<ggemm, blk, 0, stream>>>(out3, W_fr, fr_b, hbuf, 1);

    // folded [ensemble-mean conv3x3 + BN] + ReLU -> f32 d_out
    conv_gemm<false, false, 9, false, false, true>
        <<<ggemm, blk, 0, stream>>>(hbuf, W_ens, bias_f, d_out, 1);

    (void)in_sizes; (void)n_in; (void)out_size; (void)ws_size;
}
